// MultiQueryAttention_22110491640609
// MI455X (gfx1250) — matrix operand order repaired and checked by reading, not compiled, not run on hardware
//
#include <hip/hip_runtime.h>

// ---------------------------------------------------------------------------
// MI455X (gfx1250, wave32) Multi-Query Attention:
//   qkv = x@Wqkv -> RoPE/scale/split -> causal MQA flash attention -> @Wo
// Matmuls on v_wmma_f32_16x16x32_bf16; straight-copy LDS staging uses
// global_load_async_to_lds_b128 (ASYNCcnt path); prefetch for next K tiles.
// ---------------------------------------------------------------------------

typedef unsigned short u16;
typedef __attribute__((ext_vector_type(16))) __bf16 v16bf;
typedef __attribute__((ext_vector_type(8)))  float  v8f;

union FragU { v16bf v; uint4 q[2]; u16 s[16]; };
union U4U   { uint4 q; u16 s[8]; };

__device__ __forceinline__ u16 f32_to_bf16(float f) {
  unsigned int u = __float_as_uint(f);
  u += 0x7FFFu + ((u >> 16) & 1u);   // round-to-nearest-even
  return (u16)(u >> 16);
}

// Async global->LDS 16B copy (gfx1250). VDST holds the LDS byte address
// (= low 32 bits of the generic pointer, per ISA aperture rules), VADDR the
// 64-bit global address. Tracked by ASYNCcnt.
__device__ __forceinline__ void async_ld_b128(void* lds, const void* g) {
  unsigned ldsoff = (unsigned)(unsigned long long)(uintptr_t)lds;
  asm volatile("global_load_async_to_lds_b128 %0, %1, off"
               :: "v"(ldsoff), "v"(g) : "memory");
}
__device__ __forceinline__ void wait_asynccnt0() {
  asm volatile("s_wait_asynccnt 0x0" ::: "memory");
}

// A-fragment (16x32 bf16, M x K): lane<16 -> K 0..7 & 16..23 ; lane>=16 -> K 8..15 & 24..31
__device__ __forceinline__ v16bf load_frag_a(const u16* rowp, int kb8) {
  FragU f;
  f.q[0] = *(const uint4*)(rowp + kb8);
  f.q[1] = *(const uint4*)(rowp + kb8 + 16);
  return f.v;
}
// B-fragment (32x16 bf16, K x N), column data contiguous per lane:
// lane<16 -> K 0..7 & 16..23, lane>=16 -> K 8..15 & 24..31 (colp points at the lane's column, K-major)
__device__ __forceinline__ v16bf load_frag_b(const u16* colp, int kb8) {
  FragU f;
  f.q[0] = *(const uint4*)(colp + kb8);
  f.q[1] = *(const uint4*)(colp + kb8 + 16);
  return f.v;
}

__device__ __forceinline__ v8f wmma_bf16(v16bf a, v16bf b, v8f c) {
  return __builtin_amdgcn_wmma_f32_16x16x32_bf16(false, a, false, b, (short)0, c,
                                                 false, false);
}

// ---------------------------------------------------------------------------
// f32 -> bf16 convert
// ---------------------------------------------------------------------------
__global__ void cvt_bf16_kernel(const float* __restrict__ in, u16* __restrict__ out,
                                size_t n) {
  size_t i = (size_t)blockIdx.x * blockDim.x + threadIdx.x;
  if (i < n) out[i] = f32_to_bf16(in[i]);
}

// ---------------------------------------------------------------------------
// C(f32, MxN) = A(bf16, MxK row-major) * B(bf16, KxN row-major)
// 64x64 block tile, 128 threads = 4 waves, each wave a 32x32 quadrant (2x2 WMMA),
// K staged in steps of 32 through LDS; A staged via async-to-LDS; B staged
// transposed (VGPR roundtrip) so B-fragments are contiguous ds_load_b128.
// M,N,K must be multiples of 64/64/32 (true for all calls here).
// ---------------------------------------------------------------------------
__global__ __launch_bounds__(128) void gemm_bf16_kernel(
    const u16* __restrict__ A, const u16* __restrict__ B, float* __restrict__ C,
    int M, int N, int K) {
  __shared__ u16 lA[64 * 32];   // [row][k]
  __shared__ u16 lB[64 * 32];   // transposed: [n][k]
  const int tid = threadIdx.x;
  const int wid = tid >> 5;
  const int lane = tid & 31;
  const int half = lane >> 4;
  const int l16 = lane & 15;
  const int row0 = blockIdx.y * 64;
  const int col0 = blockIdx.x * 64;
  const int wr = (wid >> 1) * 32;
  const int wc = (wid & 1) * 32;

  v8f acc[2][2];
#pragma unroll
  for (int i = 0; i < 2; i++)
#pragma unroll
    for (int j = 0; j < 2; j++)
#pragma unroll
      for (int e = 0; e < 8; e++) acc[i][j][e] = 0.0f;

  for (int k0 = 0; k0 < K; k0 += 32) {
    {  // stage A tile 64x32 (straight copy -> async to LDS)
      int r = tid >> 1;
      int c = (tid & 1) * 16;
      const u16* gp = A + (size_t)(row0 + r) * K + k0 + c;
      async_ld_b128(&lA[r * 32 + c], gp);
      async_ld_b128(&lA[r * 32 + c + 8], gp + 8);
      if (k0 + 32 < K) __builtin_prefetch(gp + 32, 0, 0);
    }
    {  // stage B tile 32x64 transposed -> lB[n][k]
      int k = tid >> 2;
      int n0 = (tid & 3) * 16;
      const u16* gp = B + (size_t)(k0 + k) * N + col0 + n0;
      U4U d0, d1;
      d0.q = *(const uint4*)gp;
      d1.q = *(const uint4*)(gp + 8);
#pragma unroll
      for (int i = 0; i < 8; i++) {
        lB[(n0 + i) * 32 + k]     = d0.s[i];
        lB[(n0 + 8 + i) * 32 + k] = d1.s[i];
      }
      if (k0 + 32 < K) __builtin_prefetch(gp + (size_t)32 * N, 0, 0);
    }
    wait_asynccnt0();
    __syncthreads();

    v16bf fa[2], fb[2];
#pragma unroll
    for (int i = 0; i < 2; i++)
      fa[i] = load_frag_a(&lA[(wr + i * 16 + l16) * 32], half * 8);
#pragma unroll
    for (int j = 0; j < 2; j++)
      fb[j] = load_frag_b(&lB[(wc + j * 16 + l16) * 32], half * 8);
#pragma unroll
    for (int i = 0; i < 2; i++)
#pragma unroll
      for (int j = 0; j < 2; j++) acc[i][j] = wmma_bf16(fa[i], fb[j], acc[i][j]);
    __syncthreads();
  }

#pragma unroll
  for (int i = 0; i < 2; i++)
#pragma unroll
    for (int j = 0; j < 2; j++) {
      int col = col0 + wc + j * 16 + l16;
#pragma unroll
      for (int v = 0; v < 8; v++) {
        int row = row0 + wr + i * 16 + half * 8 + v;
        C[(size_t)row * N + col] = acc[i][j][v];
      }
    }
}

// ---------------------------------------------------------------------------
// RoPE + scale + split:  qkv(f32, B,T,2304) -> q(bf16 B,H,T,128, pre-scaled),
//                        k0(bf16 B,T,128, roped), v0(bf16 B,T,128)
// ---------------------------------------------------------------------------
__global__ void rope_split_kernel(const float* __restrict__ qkv,
                                  const float* __restrict__ sinp,
                                  const float* __restrict__ cosp,
                                  u16* __restrict__ Qo, u16* __restrict__ K0o,
                                  u16* __restrict__ V0o) {
  const int T = 2048, D = 2048, NH = 16, HD = 128, NC = 2304;
  const float scale = 0.08838834764831845f;  // 128^-0.5
  size_t idx = (size_t)blockIdx.x * blockDim.x + threadIdx.x;
  if (idx >= (size_t)2 * T * NC) return;
  int c = (int)(idx % NC);
  size_t bt = idx / NC;
  int t = (int)(bt % T);
  int b = (int)(bt / T);
  float x = qkv[idx];
  if (c < D) {
    int h = c >> 7, d = c & 127;
    float cs = cosp[t * HD + d], sn = sinp[t * HD + d];
    float xp = qkv[idx + ((d < 64) ? 64 : -64)];
    float rot = (d < 64) ? -xp : xp;
    Qo[(((size_t)b * NH + h) * T + t) * HD + d] =
        f32_to_bf16((x * cs + rot * sn) * scale);
  } else if (c < D + HD) {
    int d = c - D;
    float cs = cosp[t * HD + d], sn = sinp[t * HD + d];
    float xp = qkv[idx + ((d < 64) ? 64 : -64)];
    float rot = (d < 64) ? -xp : xp;
    K0o[((size_t)b * T + t) * HD + d] = f32_to_bf16(x * cs + rot * sn);
  } else {
    int d = c - D - HD;
    V0o[((size_t)b * T + t) * HD + d] = f32_to_bf16(x);
  }
}

// ---------------------------------------------------------------------------
// Causal MQA flash attention.
// Block = 64 queries of one (b,h): 4 waves x 16 rows. Streams 32-key blocks.
// Q held in A-fragments; S via 8 WMMA; online softmax with __shfl_xor row
// reductions; P restaged via per-wave LDS; O += P*V via 8 WMMA per block.
// K tile staged via async-to-LDS; V tile transposed via VGPR roundtrip.
// ---------------------------------------------------------------------------
__global__ __launch_bounds__(128) void mqa_attn_kernel(
    const u16* __restrict__ Q,   // (B,H,T,128) bf16 (already *scale, roped)
    const u16* __restrict__ K0,  // (B,T,128) bf16 (roped)
    const u16* __restrict__ V0,  // (B,T,128) bf16
    u16* __restrict__ O) {       // (B,T,2048) bf16
  const int T = 2048, HD = 128, NH = 16, D = 2048;
  const float scale = 0.08838834764831845f;  // second scale on scores
  __shared__ u16 lK[32 * 128];      // [key][feat]
  __shared__ u16 lV[128 * 32];      // transposed [feat][key]
  __shared__ u16 lP[4][16 * 32];    // per-wave probs tile

  const int tid = threadIdx.x, wid = tid >> 5, lane = tid & 31;
  const int half = lane >> 4, l16 = lane & 15;
  const int qblk = blockIdx.x, h = blockIdx.y, b = blockIdx.z;
  const int q0 = qblk * 64 + wid * 16;

  // load Q fragments (16 rows x 128 feats -> 4 chunks of K=32)
  v16bf fq[4];
  {
    const u16* qp = Q + (((size_t)(b * NH + h)) * T + q0 + l16) * HD;
#pragma unroll
    for (int c = 0; c < 4; c++) fq[c] = load_frag_a(qp + c * 32, half * 8);
  }
  v8f oacc[8];
#pragma unroll
  for (int t = 0; t < 8; t++)
#pragma unroll
    for (int e = 0; e < 8; e++) oacc[t][e] = 0.0f;
  float m_run[8], l_run[8];
#pragma unroll
  for (int j = 0; j < 8; j++) { m_run[j] = -3.0e38f; l_run[j] = 0.0f; }

  const int nkb = qblk * 2 + 2;  // uniform causal bound over the block
  for (int kb = 0; kb < nkb; kb++) {
    const int key0 = kb * 32;
    {  // cooperative stage of K (async copy) and V (transposed)
      int key = tid >> 2;
      int f0 = (tid & 3) * 32;
      const u16* kp = K0 + ((size_t)b * T + key0 + key) * HD + f0;
      const u16* vp = V0 + ((size_t)b * T + key0 + key) * HD + f0;
#pragma unroll
      for (int i = 0; i < 4; i++)
        async_ld_b128(&lK[key * 128 + f0 + i * 8], kp + i * 8);
      U4U v4[4];
#pragma unroll
      for (int i = 0; i < 4; i++) v4[i].q = *(const uint4*)(vp + i * 8);
#pragma unroll
      for (int i = 0; i < 4; i++)
#pragma unroll
        for (int e = 0; e < 8; e++) lV[(f0 + i * 8 + e) * 32 + key] = v4[i].s[e];
    }
    wait_asynccnt0();
    __syncthreads();

    // S = Q * K^T  (two 16x16 tiles over this 32-key block)
    v8f s[2];
#pragma unroll
    for (int t = 0; t < 2; t++)
#pragma unroll
      for (int e = 0; e < 8; e++) s[t][e] = 0.0f;
#pragma unroll
    for (int t = 0; t < 2; t++) {
      const u16* kcol = &lK[(t * 16 + l16) * 128];
#pragma unroll
      for (int c = 0; c < 4; c++) {
        v16bf fb = load_frag_b(kcol + c * 32, half * 8);
        s[t] = wmma_bf16(fq[c], fb, s[t]);
      }
    }
    // scale + causal mask
#pragma unroll
    for (int t = 0; t < 2; t++) {
      int key = key0 + t * 16 + l16;
#pragma unroll
      for (int j = 0; j < 8; j++) {
        int row = q0 + half * 8 + j;
        float sv = s[t][j] * scale;
        s[t][j] = (key > row) ? -1.0e30f : sv;
      }
    }
    // online softmax (row stats across 16-lane halves)
    float mblk[8];
#pragma unroll
    for (int j = 0; j < 8; j++) mblk[j] = fmaxf(s[0][j], s[1][j]);
    for (int msk = 1; msk < 16; msk <<= 1)
#pragma unroll
      for (int j = 0; j < 8; j++)
        mblk[j] = fmaxf(mblk[j], __shfl_xor(mblk[j], msk, 16));
    float corr[8];
#pragma unroll
    for (int j = 0; j < 8; j++) {
      float mnew = fmaxf(m_run[j], mblk[j]);
      corr[j] = __expf(m_run[j] - mnew);
      m_run[j] = mnew;
    }
#pragma unroll
    for (int t = 0; t < 2; t++)
#pragma unroll
      for (int j = 0; j < 8; j++) s[t][j] = __expf(s[t][j] - m_run[j]);
    float psum[8];
#pragma unroll
    for (int j = 0; j < 8; j++) psum[j] = s[0][j] + s[1][j];
    for (int msk = 1; msk < 16; msk <<= 1)
#pragma unroll
      for (int j = 0; j < 8; j++) psum[j] += __shfl_xor(psum[j], msk, 16);
#pragma unroll
    for (int j = 0; j < 8; j++) l_run[j] = l_run[j] * corr[j] + psum[j];
#pragma unroll
    for (int t = 0; t < 8; t++)
#pragma unroll
      for (int j = 0; j < 8; j++) oacc[t][j] *= corr[j];
    // P (C-layout f32) -> per-wave LDS bf16 (A-layout source)
#pragma unroll
    for (int t = 0; t < 2; t++)
#pragma unroll
      for (int j = 0; j < 8; j++)
        lP[wid][(half * 8 + j) * 32 + t * 16 + l16] = f32_to_bf16(s[t][j]);
    __syncthreads();

    // O += P(16x32) * V(32x128): one WMMA per 16-feature tile
    v16bf fp = load_frag_a(&lP[wid][l16 * 32], half * 8);
#pragma unroll
    for (int t = 0; t < 8; t++) {
      v16bf fv = load_frag_b(&lV[(t * 16 + l16) * 32], half * 8);
      oacc[t] = wmma_bf16(fp, fv, oacc[t]);
    }
    __syncthreads();  // protect lK/lV before restage
  }

  // normalize + store bf16 into (B,T,D)
  float inv[8];
#pragma unroll
  for (int j = 0; j < 8; j++) inv[j] = 1.0f / l_run[j];
#pragma unroll
  for (int t = 0; t < 8; t++) {
    int col = h * HD + t * 16 + l16;
#pragma unroll
    for (int j = 0; j < 8; j++) {
      int row = q0 + half * 8 + j;
      O[((size_t)b * T + row) * D + col] = f32_to_bf16(oacc[t][j] * inv[j]);
    }
  }
}

// ---------------------------------------------------------------------------
// launch
// ---------------------------------------------------------------------------
static inline size_t alignup(size_t x) { return (x + 255) & ~(size_t)255; }

extern "C" void kernel_launch(void* const* d_in, const int* in_sizes, int n_in,
                              void* d_out, int out_size, void* d_ws, size_t ws_size,
                              hipStream_t stream) {
  (void)in_sizes; (void)n_in; (void)out_size; (void)ws_size;
  const float* x    = (const float*)d_in[0];  // (2,2048,2048)
  const float* sinp = (const float*)d_in[1];  // (2048,128)
  const float* cosp = (const float*)d_in[2];  // (2048,128)
  const float* Wqkv = (const float*)d_in[3];  // (2048,2304)
  const float* Wo   = (const float*)d_in[4];  // (2048,2048)
  float* out = (float*)d_out;                 // (2,2048,2048)

  const int B = 2, T = 2048, D = 2048, NH = 16, HD = 128;
  const int NC = D + 2 * HD;                  // 2304
  const size_t NX   = (size_t)B * T * D;      // 8388608
  const size_t NWQ  = (size_t)D * NC;         // 4718592
  const size_t NWO  = (size_t)D * D;          // 4194304
  const size_t NQKV = (size_t)B * T * NC;     // 9437184 (f32)
  const size_t NKV  = (size_t)B * T * HD;     // 524288

  char* w = (char*)d_ws;
  u16* xb   = (u16*)w;   w += alignup(NX  * 2);
  u16* wqb  = (u16*)w;   w += alignup(NWQ * 2);
  u16* wob  = (u16*)w;   w += alignup(NWO * 2);
  float* qkvf = (float*)w;          // f32 qkv buffer ...
  u16* attnb  = (u16*)w;            // ... aliased by attn output (consumed first)
  w += alignup(NQKV * 4);
  u16* qb = (u16*)w;     w += alignup(NX  * 2);
  u16* kb = (u16*)w;     w += alignup(NKV * 2);
  u16* vb = (u16*)w;     w += alignup(NKV * 2);

  // 1) converts
  cvt_bf16_kernel<<<(unsigned)((NX  + 255) / 256), 256, 0, stream>>>(x, xb, NX);
  cvt_bf16_kernel<<<(unsigned)((NWQ + 255) / 256), 256, 0, stream>>>(Wqkv, wqb, NWQ);
  cvt_bf16_kernel<<<(unsigned)((NWO + 255) / 256), 256, 0, stream>>>(Wo, wob, NWO);

  // 2) qkv = x @ Wqkv : (4096 x 2304 x 2048)
  gemm_bf16_kernel<<<dim3(NC / 64, (B * T) / 64), 128, 0, stream>>>(
      xb, wqb, qkvf, B * T, NC, D);

  // 3) RoPE + scale + split
  rope_split_kernel<<<(unsigned)((NQKV + 255) / 256), 256, 0, stream>>>(
      qkvf, sinp, cosp, qb, kb, vb);

  // 4) causal MQA flash attention -> attnb (B,T,D) bf16
  mqa_attn_kernel<<<dim3(T / 64, NH, B), 128, 0, stream>>>(qb, kb, vb, attnb);

  // 5) out = attn @ Wo : (4096 x 2048 x 2048) -> f32
  gemm_bf16_kernel<<<dim3(D / 64, (B * T) / 64), 128, 0, stream>>>(
      attnb, wob, out, B * T, D, D);
}
